// HGTEncoder_20426864460188
// MI455X (gfx1250) — matrix operand order repaired and checked by reading, not compiled, not run on hardware
//
#include <hip/hip_runtime.h>
#include <math.h>

#define NP_   100000
#define NA_   50000
#define NTOT  (NP_ + NA_)
#define E0_   300000
#define E1_   300000
#define E2_   200000
#define ETOT  (E0_ + E1_ + E2_)
#define SRCTOT (NA_ + NP_ + NP_)
#define MAXK  128

typedef __attribute__((ext_vector_type(16))) _Float16 v16h;
typedef __attribute__((ext_vector_type(8)))  _Float16 v8h;
typedef __attribute__((ext_vector_type(8)))  float    v8f;

__device__ __forceinline__ float gelu_exact(float x) {
  return 0.5f * x * (1.0f + erff(x * 0.70710678118654752f));
}

__device__ __forceinline__ void atomicMaxF(float* addr, float v) {
  if (v >= 0.0f) atomicMax((int*)addr, __float_as_int(v));
  else           atomicMin((unsigned int*)addr, (unsigned int)__float_as_int(v));
}

// ---------------------------------------------------------------------------
// WMMA GEMM: C[M,N] = A[M,K] @ B[K,N] + bias  (f32 in/out, f16 MACs)
// B slice [K,16] for this block's n-tile is staged once into LDS as f16
// (padded row stride K+8 halfs -> conflict-free ds_load_b128 reads).
// mode 0: plain store; mode 1: C = s*val + (1-s)*resid, s = sigmoid(*skipp)
// One wave computes one 16x16 tile; 4 waves/block over M.
// ---------------------------------------------------------------------------
__global__ void __launch_bounds__(128)
wmma_gemm(const float* __restrict__ A, int lda,
          const float* __restrict__ B, int ldb,
          const float* __restrict__ bias,
          float* __restrict__ C, int ldc,
          int M, int K, int mode,
          const float* __restrict__ resid,
          const float* __restrict__ skipp)
{
  __shared__ __align__(32) _Float16 lb[16 * (MAXK + 8)];
  const int tid = threadIdx.x;
  const int n0  = blockIdx.y * 16;
  const int strideh = K + 8;                 // halfs; row stride is 16B multiple

  // cooperative stage: B[k][n0+col] -> lb[col][k] as f16 (coalesced b32 loads)
  for (int idx = tid; idx < 16 * K; idx += 128) {
    const int col = idx & 15;
    const int k   = idx >> 4;
    lb[col * strideh + k] = (_Float16)B[(size_t)k * ldb + n0 + col];
  }
  __syncthreads();

  const int lane = tid & 31;
  const int wv   = tid >> 5;
  const int mt   = blockIdx.x * 4 + wv;
  if (mt * 16 >= M) return;                  // uniform per wave, after barrier
  const int m0  = mt * 16;
  const int col = lane & 15;                 // also A row-in-tile
  const int hi  = lane >> 4;

  v8f acc = {};
  const float* Arow = A + (size_t)(m0 + col) * lda;
  for (int k0 = 0; k0 < K; k0 += 32) {
    __builtin_prefetch(Arow + k0 + 32, 0, 3);   // global_prefetch_b8 (speculative)
    // A: ISA 16x32 f16 layout = two 8-float contiguous runs per lane
    const float* p0 = Arow + k0 + hi * 8;
    const float* p1 = p0 + 16;
    const float4 x0 = *(const float4*)p0, x1 = *(const float4*)(p0 + 4);
    const float4 x2 = *(const float4*)p1, x3 = *(const float4*)(p1 + 4);
    v16h a;
    a[0]  = (_Float16)x0.x; a[1]  = (_Float16)x0.y; a[2]  = (_Float16)x0.z; a[3]  = (_Float16)x0.w;
    a[4]  = (_Float16)x1.x; a[5]  = (_Float16)x1.y; a[6]  = (_Float16)x1.z; a[7]  = (_Float16)x1.w;
    a[8]  = (_Float16)x2.x; a[9]  = (_Float16)x2.y; a[10] = (_Float16)x2.z; a[11] = (_Float16)x2.w;
    a[12] = (_Float16)x3.x; a[13] = (_Float16)x3.y; a[14] = (_Float16)x3.z; a[15] = (_Float16)x3.w;
    // B: ISA 32x16 f16 layout = two 8-half runs (k 8h.., 16+8h..) from LDS (2x ds_load_b128)
    const _Float16* bp = &lb[col * strideh + k0 + hi * 8];
    const v8h b0 = *(const v8h*)bp;
    const v8h b1 = *(const v8h*)(bp + 16);
    v16h b;
#pragma unroll
    for (int i = 0; i < 8; ++i) { b[i] = b0[i]; b[8 + i] = b1[i]; }
    acc = __builtin_amdgcn_wmma_f32_16x16x32_f16(false, a, false, b,
                                                 (short)0, acc, false, false);
  }

  const int n = n0 + col;
  const float bv = bias ? bias[n] : 0.0f;
  float sg = 0.0f;
  if (mode == 1) sg = 1.0f / (1.0f + expf(-skipp[0]));
#pragma unroll
  for (int r = 0; r < 8; ++r) {              // ISA C/D f32 layout: M = r + 8*hi
    const int m = m0 + r + hi * 8;
    float v = acc[r] + bv;
    if (mode == 1) v = sg * v + (1.0f - sg) * resid[(size_t)m * ldc + n];
    C[(size_t)m * ldc + n] = v;
  }
}

// ---------------------------------------------------------------------------
// Per-head relation transform: Ks[n,h,:] = k[n,h,:] @ Wk[h], same for V.
// K=16 zero-padded to 32 so it runs on v_wmma_f32_16x16x32_f16.
// Block = 4 waves = 4 heads of one 16-node tile.
// ---------------------------------------------------------------------------
__global__ void __launch_bounds__(128)
rel_transform(const float* __restrict__ kqv_src,  // [n,192]; k at +0, v at +128
              const float* __restrict__ Wk,       // [4,16,16]
              const float* __restrict__ Wv,       // [4,16,16]
              float* __restrict__ Ks,             // [n,64] (pre-offset)
              float* __restrict__ Vs)             // [n,64]
{
  const int lane = threadIdx.x & 31;
  const int h    = threadIdx.x >> 5;
  const int m0   = blockIdx.x * 16;
  const int col  = lane & 15;
  const int hi   = lane >> 4;

  const float* krow = kqv_src + (size_t)(m0 + col) * 192 + h * 16 + hi * 8;
  const float* vrow = krow + 128;
  v16h ak = {}, av = {}, bk = {}, bv = {};
  {
    const float4 k0 = *(const float4*)krow, k1 = *(const float4*)(krow + 4);
    const float4 v0 = *(const float4*)vrow, v1 = *(const float4*)(vrow + 4);
    ak[0] = (_Float16)k0.x; ak[1] = (_Float16)k0.y; ak[2] = (_Float16)k0.z; ak[3] = (_Float16)k0.w;
    ak[4] = (_Float16)k1.x; ak[5] = (_Float16)k1.y; ak[6] = (_Float16)k1.z; ak[7] = (_Float16)k1.w;
    av[0] = (_Float16)v0.x; av[1] = (_Float16)v0.y; av[2] = (_Float16)v0.z; av[3] = (_Float16)v0.w;
    av[4] = (_Float16)v1.x; av[5] = (_Float16)v1.y; av[6] = (_Float16)v1.z; av[7] = (_Float16)v1.w;
  }
  {                                         // B elements 8..15 (K=16..31) stay zero
    const float* Wkh = Wk + h * 256 + col + hi * 8 * 16;
    const float* Wvh = Wv + h * 256 + col + hi * 8 * 16;
#pragma unroll
    for (int i = 0; i < 4; ++i) {
      bk[2*i]   = (_Float16)Wkh[(2*i)     * 16];
      bk[2*i+1] = (_Float16)Wkh[(2*i + 1) * 16];
      bv[2*i]   = (_Float16)Wvh[(2*i)     * 16];
      bv[2*i+1] = (_Float16)Wvh[(2*i + 1) * 16];
    }
  }
  v8f ck = {}, cv = {};
  ck = __builtin_amdgcn_wmma_f32_16x16x32_f16(false, ak, false, bk, (short)0, ck, false, false);
  cv = __builtin_amdgcn_wmma_f32_16x16x32_f16(false, av, false, bv, (short)0, cv, false, false);
#pragma unroll
  for (int r = 0; r < 8; ++r) {
    const size_t row = (size_t)(m0 + r + hi * 8);
    Ks[row * 64 + h * 16 + col] = ck[r];
    Vs[row * 64 + h * 16 + col] = cv[r];
  }
}

// ---------------------------------------------------------------------------
// Edge attention pass 1: raw score + atomic segment max
// ---------------------------------------------------------------------------
__global__ void __launch_bounds__(256)
attn_scores(const float* __restrict__ kqv, const float* __restrict__ Ks,
            const int* __restrict__ src, const int* __restrict__ dst,
            int E, int src_base, int dst_base,
            const float* __restrict__ prel,
            float* __restrict__ alpha, float* __restrict__ segmax)
{
  const int t = blockIdx.x * blockDim.x + threadIdx.x;
  const int e = t >> 2, h = t & 3;
  if (e >= E) return;
  const int s = src[e] + src_base;
  const int d = dst[e] + dst_base;
  const float4* q4 = (const float4*)(kqv + (size_t)d * 192 + 64 + h * 16);
  const float4* k4 = (const float4*)(Ks  + (size_t)s * 64 + h * 16);
  float acc = 0.0f;
#pragma unroll
  for (int i = 0; i < 4; ++i) {
    const float4 q = q4[i], k = k4[i];
    acc = fmaf(q.x, k.x, acc); acc = fmaf(q.y, k.y, acc);
    acc = fmaf(q.z, k.z, acc); acc = fmaf(q.w, k.w, acc);
  }
  const float a = acc * prel[h] * 0.25f;    // /sqrt(16)
  alpha[(size_t)e * 4 + h] = a;
  atomicMaxF(&segmax[d * 4 + h], a);
}

// pass 2: exp(a - max), atomic segment sum
__global__ void __launch_bounds__(256)
attn_expsum(const int* __restrict__ dst, int E, int dst_base,
            float* __restrict__ alpha, const float* __restrict__ segmax,
            float* __restrict__ segsum)
{
  const int t = blockIdx.x * blockDim.x + threadIdx.x;
  const int e = t >> 2, h = t & 3;
  if (e >= E) return;
  const int d = dst[e] + dst_base;
  const float ex = expf(alpha[(size_t)e * 4 + h] - segmax[d * 4 + h]);
  alpha[(size_t)e * 4 + h] = ex;
  atomicAdd(&segsum[d * 4 + h], ex);
}

// pass 3: agg[dst] += alpha * Vs[src]
__global__ void __launch_bounds__(256)
attn_aggregate(const float* __restrict__ Vs,
               const int* __restrict__ src, const int* __restrict__ dst,
               int E, int src_base, int dst_base,
               const float* __restrict__ alpha, float* __restrict__ agg)
{
  const int t = blockIdx.x * blockDim.x + threadIdx.x;
  const int e = t >> 2, h = t & 3;
  if (e >= E) return;
  const int s = src[e] + src_base;
  const int d = dst[e] + dst_base;
  const float w = alpha[(size_t)e * 4 + h];
  const float4* v4 = (const float4*)(Vs + (size_t)s * 64 + h * 16);
  float* out = agg + (size_t)d * 64 + h * 16;
#pragma unroll
  for (int i = 0; i < 4; ++i) {
    const float4 v = v4[i];
    atomicAdd(out + 4*i + 0, v.x * w);
    atomicAdd(out + 4*i + 1, v.y * w);
    atomicAdd(out + 4*i + 2, v.z * w);
    atomicAdd(out + 4*i + 3, v.w * w);
  }
}

// softmax denominator + GELU (in place on agg)
__global__ void __launch_bounds__(256)
norm_gelu(const float* __restrict__ segsum, float* __restrict__ agg)
{
  const int t = blockIdx.x * blockDim.x + threadIdx.x;
  if (t >= NTOT * 64) return;
  const int n = t >> 6, i = t & 63;
  const float denom = segsum[n * 4 + (i >> 4)] + 1e-16f;
  agg[t] = gelu_exact(agg[t] / denom);
}

// LayerNorm(64) + GELU, one wave32 per row (2 cols/lane), in place
__global__ void __launch_bounds__(256)
ln_gelu(float* __restrict__ hbuf, const float* __restrict__ lg,
        const float* __restrict__ lb)
{
  const int lane = threadIdx.x & 31;
  const int wv   = threadIdx.x >> 5;
  const int n    = blockIdx.x * 8 + wv;
  if (n >= NTOT) return;
  float* row = hbuf + (size_t)n * 64;
  const float x0 = row[lane], x1 = row[lane + 32];
  float s = x0 + x1;
  for (int off = 16; off > 0; off >>= 1) s += __shfl_xor(s, off, 32);
  const float mean = s * (1.0f / 64.0f);
  const float d0 = x0 - mean, d1 = x1 - mean;
  float v = d0 * d0 + d1 * d1;
  for (int off = 16; off > 0; off >>= 1) v += __shfl_xor(v, off, 32);
  const float rstd = rsqrtf(v * (1.0f / 64.0f) + 1e-5f);
  const int to = (n < NP_) ? 0 : 64;        // per-type LN params
  row[lane]      = gelu_exact(d0 * rstd * lg[to + lane]      + lb[to + lane]);
  row[lane + 32] = gelu_exact(d1 * rstd * lg[to + lane + 32] + lb[to + lane + 32]);
}

__global__ void __launch_bounds__(256)
init_seg(float* __restrict__ segmax, float* __restrict__ segsum,
         float* __restrict__ agg)
{
  const int t = blockIdx.x * blockDim.x + threadIdx.x;
  if (t < NTOT * 4) { segmax[t] = -INFINITY; segsum[t] = 0.0f; }
  if (t < NTOT * 64) agg[t] = 0.0f;
}

// ---------------------------------------------------------------------------
extern "C" void kernel_launch(void* const* d_in, const int* in_sizes, int n_in,
                              void* d_out, int out_size, void* d_ws, size_t ws_size,
                              hipStream_t stream)
{
  (void)in_sizes; (void)n_in; (void)out_size; (void)ws_size;
  const float* x_p    = (const float*)d_in[0];
  const float* x_a    = (const float*)d_in[1];
  const int*   ei_ap  = (const int*)d_in[2];
  const int*   ei_pa  = (const int*)d_in[3];
  const int*   ei_pp  = (const int*)d_in[4];
  const float* W_in   = (const float*)d_in[5];
  const float* b_in   = (const float*)d_in[6];
  const float* W_kqv  = (const float*)d_in[7];
  const float* b_kqv  = (const float*)d_in[8];
  const float* W_krel = (const float*)d_in[9];
  const float* W_vrel = (const float*)d_in[10];
  const float* p_rel  = (const float*)d_in[11];
  const float* W_hout = (const float*)d_in[12];
  const float* b_hout = (const float*)d_in[13];
  const float* skip   = (const float*)d_in[14];
  const float* ln_g   = (const float*)d_in[15];
  const float* ln_b   = (const float*)d_in[16];
  const float* W_out  = (const float*)d_in[17];
  const float* b_out  = (const float*)d_in[18];
  float* out = (float*)d_out;

  // workspace carve (f32): h | kqv | Ks | Vs | alpha | segmax | segsum | agg
  float* hbuf   = (float*)d_ws;
  float* kqv    = hbuf   + (size_t)NTOT * 64;
  float* Ks     = kqv    + (size_t)NTOT * 192;
  float* Vs     = Ks     + (size_t)SRCTOT * 64;
  float* alpha  = Vs     + (size_t)SRCTOT * 64;
  float* segmax = alpha  + (size_t)ETOT * 4;
  float* segsum = segmax + (size_t)NTOT * 4;
  float* agg    = segsum + (size_t)NTOT * 4;

  auto gemm = [&](const float* A, int lda, const float* B, int ldb,
                  const float* bias, float* C, int ldc,
                  int M, int K, int N, int mode,
                  const float* resid, const float* skp) {
    dim3 grid((M / 16 + 3) / 4, N / 16);
    wmma_gemm<<<grid, 128, 0, stream>>>(A, lda, B, ldb, bias, C, ldc,
                                        M, K, mode, resid, skp);
  };

  // input projections -> hbuf
  gemm(x_p, 128, W_in,            64, b_in,      hbuf,                    64, NP_, 128, 64, 0, nullptr, nullptr);
  gemm(x_a, 128, W_in + 128 * 64, 64, b_in + 64, hbuf + (size_t)NP_ * 64, 64, NA_, 128, 64, 0, nullptr, nullptr);

  const int* src_[3] = { ei_ap,       ei_pa,       ei_pp       };
  const int* dst_[3] = { ei_ap + E0_, ei_pa + E1_, ei_pp + E2_ };
  const int  Ee_[3]  = { E0_, E1_, E2_ };
  const int  sb_[3]  = { 0, NA_, NA_ + NP_ };
  const int  db_[3]  = { 0, NP_, 0 };
  const size_t ao_[3] = { 0, (size_t)E0_, (size_t)(E0_ + E1_) };

  for (int l = 0; l < 2; ++l) {
    // fused KQV projections
    gemm(hbuf, 64, W_kqv + (size_t)(l * 2 + 0) * 64 * 192, 192,
         b_kqv + (l * 2 + 0) * 192, kqv, 192, NP_, 64, 192, 0, nullptr, nullptr);
    gemm(hbuf + (size_t)NP_ * 64, 64, W_kqv + (size_t)(l * 2 + 1) * 64 * 192, 192,
         b_kqv + (l * 2 + 1) * 192, kqv + (size_t)NP_ * 192, 192, NA_, 64, 192, 0, nullptr, nullptr);

    // per-edge-type relation transforms (K and V fused per wave)
    rel_transform<<<NA_ / 16, 128, 0, stream>>>(
        kqv + (size_t)NP_ * 192,
        W_krel + (size_t)(l * 3 + 0) * 1024, W_vrel + (size_t)(l * 3 + 0) * 1024,
        Ks, Vs);
    rel_transform<<<NP_ / 16, 128, 0, stream>>>(
        kqv,
        W_krel + (size_t)(l * 3 + 1) * 1024, W_vrel + (size_t)(l * 3 + 1) * 1024,
        Ks + (size_t)NA_ * 64, Vs + (size_t)NA_ * 64);
    rel_transform<<<NP_ / 16, 128, 0, stream>>>(
        kqv,
        W_krel + (size_t)(l * 3 + 2) * 1024, W_vrel + (size_t)(l * 3 + 2) * 1024,
        Ks + (size_t)(NA_ + NP_) * 64, Vs + (size_t)(NA_ + NP_) * 64);

    init_seg<<<(NTOT * 64 + 255) / 256, 256, 0, stream>>>(segmax, segsum, agg);

    for (int i = 0; i < 3; ++i) {
      const int blocks = (Ee_[i] * 4 + 255) / 256;
      attn_scores<<<blocks, 256, 0, stream>>>(kqv, Ks, src_[i], dst_[i], Ee_[i],
                                              sb_[i], db_[i],
                                              p_rel + (l * 3 + i) * 4,
                                              alpha + ao_[i] * 4, segmax);
    }
    for (int i = 0; i < 3; ++i) {
      const int blocks = (Ee_[i] * 4 + 255) / 256;
      attn_expsum<<<blocks, 256, 0, stream>>>(dst_[i], Ee_[i], db_[i],
                                              alpha + ao_[i] * 4, segmax, segsum);
    }
    for (int i = 0; i < 3; ++i) {
      const int blocks = (Ee_[i] * 4 + 255) / 256;
      attn_aggregate<<<blocks, 256, 0, stream>>>(Vs, src_[i], dst_[i], Ee_[i],
                                                 sb_[i], db_[i],
                                                 alpha + ao_[i] * 4, agg);
    }

    norm_gelu<<<(NTOT * 64 + 255) / 256, 256, 0, stream>>>(segsum, agg);

    // hout + sigmoid skip-gate (fused epilogue), writes hbuf in place
    gemm(agg, 64, W_hout + (size_t)(l * 2 + 0) * 4096, 64,
         b_hout + (l * 2 + 0) * 64, hbuf, 64, NP_, 64, 64,
         1, hbuf, skip + l * 2 + 0);
    gemm(agg + (size_t)NP_ * 64, 64, W_hout + (size_t)(l * 2 + 1) * 4096, 64,
         b_hout + (l * 2 + 1) * 64, hbuf + (size_t)NP_ * 64, 64, NA_, 64, 64,
         1, hbuf + (size_t)NP_ * 64, skip + l * 2 + 1);

    ln_gelu<<<NTOT / 8, 256, 0, stream>>>(hbuf, ln_g + l * 128, ln_b + l * 128);
  }

  // output projections -> d_out
  gemm(hbuf, 64, W_out, 64, b_out, out, 64, NP_, 64, 64, 0, nullptr, nullptr);
  gemm(hbuf + (size_t)NP_ * 64, 64, W_out + 4096, 64, b_out + 64,
       out + (size_t)NP_ * 64, 64, NA_, 64, 64, 0, nullptr, nullptr);
}
